// _CCHead_85495618994775
// MI455X (gfx1250) — matrix operand order repaired and checked by reading, not compiled, not run on hardware
//
#include <hip/hip_runtime.h>
#include <math.h>

typedef __attribute__((ext_vector_type(16))) __bf16 v16bf;
typedef __attribute__((ext_vector_type(8)))  __bf16 v8bf;
typedef __attribute__((ext_vector_type(8)))  float  v8f;
typedef __attribute__((ext_vector_type(4)))  unsigned int u32x4;
typedef __attribute__((ext_vector_type(8)))  int   i32x8;
typedef __attribute__((ext_vector_type(4)))  int   i32x4;

typedef unsigned short ushort_t;

#define BB   2
#define HH   96
#define WW   96
#define HWC  (HH*WW)          // 9216
#define BHW  (BB*HWC)         // 18432
#define CIN  2048
#define CI   512
#define CQK  64
#define NCLS 19

#define PH   98               // padded height  (rows -1..96 -> 0..97)
#define PW   104              // padded width, multiple of 8 (cols stored at w+4)
#define PHW  (PH*PW)

#define BM   128
#define BN   96               // one full output row (6 x 16)
#define BK   32
#define LSTR 40               // LDS K-row stride in ushorts (80B = 5*16B)

#if __has_builtin(__builtin_amdgcn_tensor_load_to_lds) && __has_builtin(__builtin_amdgcn_s_wait_tensorcnt)
#define HAS_TDM 1
#else
#define HAS_TDM 0
#endif

__device__ __forceinline__ ushort_t f2bf(float f) {
    unsigned int u = __builtin_bit_cast(unsigned int, f);
    unsigned int r = u + 0x7FFFu + ((u >> 16) & 1u);
    return (ushort_t)(r >> 16);
}
__device__ __forceinline__ float bf2f(ushort_t h) {
    unsigned int u = ((unsigned int)h) << 16;
    return __builtin_bit_cast(float, u);
}

// -------------------- utility kernels --------------------

__global__ void zero_u16_k(ushort_t* __restrict__ p, size_t n) {
    size_t i = (size_t)blockIdx.x * blockDim.x + threadIdx.x;
    size_t stride = (size_t)gridDim.x * blockDim.x;
    for (; i < n; i += stride) p[i] = 0;
}

// x (B,C,H,W) f32 -> padded bf16 (B,C,PH,PW), borders zero
__global__ void cast_pad_k(const float* __restrict__ src, ushort_t* __restrict__ dst,
                           int C, size_t n) {
    size_t i = (size_t)blockIdx.x * blockDim.x + threadIdx.x;
    if (i >= n) return;
    int pcol = (int)(i % PW);
    int prow = (int)((i / PW) % PH);
    size_t bc = i / PHW;               // b*C + c
    ushort_t v = 0;
    int hh = prow - 1, ww = pcol - 4;
    if (hh >= 0 && hh < HH && ww >= 0 && ww < WW)
        v = f2bf(src[bc * HWC + (size_t)hh * WW + ww]);
    dst[i] = v;
}

// src layout (Cout, Cin, Kpos) f32 -> wt2 layout (Kpos, Cout, Cin) bf16 (K contiguous)
__global__ void cast_transpose_w_k(const float* __restrict__ w, ushort_t* __restrict__ wt,
                                   int Cout, int Cin, int Kpos, int n) {
    int i = blockIdx.x * blockDim.x + threadIdx.x;
    if (i >= n) return;
    int o   = i / (Cin * Kpos);
    int rem = i % (Cin * Kpos);
    int c   = rem / Kpos;
    int p   = rem % Kpos;
    wt[((size_t)p * Cout + o) * Cin + c] = f2bf(w[i]);
}

// -------------------- WMMA implicit-GEMM conv --------------------

__device__ __forceinline__ v16bf ldfrag(const ushort_t* rowbase, int kA, int kB) {
    v8bf lo = *reinterpret_cast<const v8bf*>(rowbase + kA);
    v8bf hi = *reinterpret_cast<const v8bf*>(rowbase + kB);
    return __builtin_shufflevector(lo, hi, 0,1,2,3,4,5,6,7,8,9,10,11,12,13,14,15);
}

template<int KPOS>
__global__ __launch_bounds__(256)
void conv_gemm_wmma_k(const ushort_t* __restrict__ in1p, int Cin1,   // padded (B,Cin1,PH,PW)
                      const ushort_t* __restrict__ in2p, int Cin2,   // virtual concat tail
                      const ushort_t* __restrict__ wt2,              // [KPOS][Cout][CinTot] bf16
                      int Cout,
                      const float* __restrict__ scale,
                      const float* __restrict__ bias,
                      int relu, int out_pad,
                      ushort_t* __restrict__ outb,
                      float* __restrict__ outf)
{
    // A tiles (double buffered): [m 0..127][k 0..31], row stride LSTR (= TDM pad layout)
    __shared__ __attribute__((aligned(16))) ushort_t lds_a[2][BM * LSTR];
    // B tile: [dh][pcol 0..103][k 0..31]
    __shared__ __attribute__((aligned(16))) ushort_t lds_b[3 * PW * LSTR];

    const int t      = threadIdx.x;
    const int lane   = t & 31;
    const int wid    = t >> 5;
    const int wave_m = (wid >> 1) * 32;   // 0,32,64,96
    const int wave_n = (wid & 1)  * 48;   // 0,48
    const int b      = blockIdx.x / HH;
    const int h      = blockIdx.x % HH;
    const int m0     = blockIdx.y * BM;
    const int CinTot = Cin1 + Cin2;

    v8f zero;
#pragma unroll
    for (int i = 0; i < 8; ++i) zero[i] = 0.0f;
    v8f acc[2][3];
#pragma unroll
    for (int mt = 0; mt < 2; ++mt)
#pragma unroll
        for (int nt = 0; nt < 3; ++nt) acc[mt][nt] = zero;

    const int ka    = (lane < 16) ? 0 : 8;    // A: K halves {0-7,16-23}/{8-15,24-31}
    const int kb    = (lane < 16) ? 0 : 8;    // B: K halves {0-7,16-23}/{8-15,24-31}
    const int arow0 = wave_m + (lane & 15);
    const int arow1 = arow0 + 16;

#if HAS_TDM
    // ---- loop-invariant TDM descriptor pieces ----
    const unsigned int ldsA0 = (unsigned int)(unsigned long long)(const void*)&lds_a[0][0];
    const unsigned int ldsA1 = (unsigned int)(unsigned long long)(const void*)&lds_a[1][0];
    const int td1 = Cout - m0;                // rows remaining; TDM zero-fills OOB rows
    i32x8 g1;
    g1[0] = (int)((1u << 16)                  // data_size = 2B
                | (1u << 20)                  // pad_enable
                | (3u << 22)                  // pad_interval: 16 dwords
                | (3u << 25));                // pad_amount: 4 dwords -> LDS row stride 80B
    g1[1] = (int)(((unsigned)CinTot & 0xffffu) << 16);                   // tensor_dim0 lo
    g1[2] = (int)((((unsigned)CinTot >> 16) & 0xffffu)
                | (((unsigned)td1 & 0xffffu) << 16));                    // dim0 hi | dim1 lo
    g1[3] = (int)((((unsigned)td1 >> 16) & 0xffffu) | (32u << 16));      // dim1 hi | tile_dim0
    g1[4] = (int)(128u);                                                 // tile_dim1=128, tile_dim2=0
    g1[5] = (int)(unsigned)CinTot;                                       // dim0_stride lo
    g1[6] = 0;
    g1[7] = 0;
    const i32x4 gz = {0, 0, 0, 0};

    auto issue_a = [&](int kpos, int c0, int bufIdx) {
        unsigned long long gaddr = (unsigned long long)(const void*)
            &wt2[((size_t)kpos * Cout + m0) * CinTot + c0];
        u32x4 g0;
        g0[0] = 1u;                                                      // count=1
        g0[1] = bufIdx ? ldsA1 : ldsA0;                                  // lds_addr
        g0[2] = (unsigned int)gaddr;
        g0[3] = ((unsigned int)(gaddr >> 32) & 0x01FFFFFFu) | 0x80000000u; // hi | type=2
#if defined(__clang_major__) && (__clang_major__ >= 23)
        i32x8 gz8 = {0, 0, 0, 0, 0, 0, 0, 0};
        __builtin_amdgcn_tensor_load_to_lds(g0, g1, gz, gz, gz8, 0);
#else
        __builtin_amdgcn_tensor_load_to_lds(g0, g1, gz, gz, 0);
#endif
    };

    if (wid == 0) issue_a(0, 0, 0);           // prefetch first A tile
#endif

    constexpr int CHUNKS = (KPOS == 9) ? 1248 : 416;   // 16B segments per B tile
    constexpr int FULLIT = CHUNKS / 256;
    constexpr int REMCNT = CHUNKS % 256;

    int tI = 0;                                // global tile index (chunk*KPOS + kpos)
    for (int c0 = 0; c0 < CinTot; c0 += BK) {
        // ---- load B tile: contiguous aligned 16B segments, transpose-store to [pcol][k] ----
#pragma unroll
        for (int it = 0; it <= FULLIT; ++it) {
            int e = it * 256 + t;
            if (it < FULLIT || (REMCNT && t < REMCNT)) {
                int dh  = (KPOS == 9) ? (e / 416) : 1;
                int r   = (KPOS == 9) ? (e % 416) : e;
                int kc  = r / 13;
                int seg = r % 13;
                int c   = c0 + kc;
                const ushort_t* srcp = (c < Cin1)
                    ? &in1p[(((size_t)b * Cin1 + c)        * PH + (h + dh)) * PW + seg * 8]
                    : &in2p[(((size_t)b * Cin2 + (c-Cin1)) * PH + (h + dh)) * PW + seg * 8];
                if (c0 + BK < CinTot)
                    __builtin_prefetch(srcp + (size_t)BK * PHW, 0, 1);
                u32x4 d = *reinterpret_cast<const u32x4*>(srcp);
                int base = (dh * PW + seg * 8) * LSTR + kc;
#pragma unroll
                for (int j = 0; j < 8; ++j) {
                    unsigned int wdw = d[j >> 1];
                    lds_b[base + j * LSTR] = (ushort_t)((j & 1) ? (wdw >> 16) : (wdw & 0xffff));
                }
            }
        }

#pragma unroll
        for (int kpos = 0; kpos < KPOS; ++kpos) {
            const int buf = tI & 1;

#if HAS_TDM
            if (wid == 0) __builtin_amdgcn_s_wait_tensorcnt(0);   // A[buf] landed
#else
            // fallback: guarded scalar loads, same [m][k] order as wt2
#pragma unroll
            for (int i = 0; i < 16; ++i) {
                int e  = i * 256 + t;
                int kk = e >> 7;
                int m  = e & 127;
                ushort_t vA = 0;
                if (m0 + m < Cout)
                    vA = wt2[((size_t)kpos * Cout + m0 + m) * CinTot + c0 + kk];
                lds_a[buf][m * LSTR + kk] = vA;
            }
#endif
            __syncthreads();   // publish A[buf] (+ B tile on first tap of chunk)

#if HAS_TDM
            {   // pipeline: issue next A tile into the other buffer
                int nk = kpos + 1, nc = c0;
                if (nk == KPOS) { nk = 0; nc = c0 + BK; }
                if (wid == 0 && nc < CinTot) issue_a(nk, nc, (tI + 1) & 1);
            }
#endif
            constexpr int taps = KPOS;  (void)taps;
            const int dhI   = (KPOS == 9) ? (kpos / 3) : 1;        // constants after unroll
            const int dwOff = (KPOS == 9) ? (kpos % 3) + 3 : 4;

            v16bf a0 = ldfrag(&lds_a[buf][arow0 * LSTR], ka, ka + 16);
            v16bf a1 = ldfrag(&lds_a[buf][arow1 * LSTR], ka, ka + 16);

            v16bf bfr[3];
#pragma unroll
            for (int nt = 0; nt < 3; ++nt) {
                int pcol = wave_n + nt * 16 + (lane & 15) + dwOff;
                bfr[nt] = ldfrag(&lds_b[(dhI * PW + pcol) * LSTR], kb, kb + 16);
            }

#pragma unroll
            for (int nt = 0; nt < 3; ++nt) {
                acc[0][nt] = __builtin_amdgcn_wmma_f32_16x16x32_bf16(false, a0, false, bfr[nt], (short)0, acc[0][nt], false, false);
                acc[1][nt] = __builtin_amdgcn_wmma_f32_16x16x32_bf16(false, a1, false, bfr[nt], (short)0, acc[1][nt], false, false);
            }
            ++tI;
        }
        __syncthreads();       // B reads done before next chunk's B stores
    }

    // ---- epilogue: VGPR r -> M=r (lanes 0-15) / M=8+r (lanes 16-31); N = lane&15 ----
#pragma unroll
    for (int mt = 0; mt < 2; ++mt)
#pragma unroll
        for (int nt = 0; nt < 3; ++nt)
#pragma unroll
            for (int r = 0; r < 8; ++r) {
                int co = m0 + wave_m + mt * 16 + r + ((lane < 16) ? 0 : 8);
                int n  = wave_n + nt * 16 + (lane & 15);    // 0..95 (= w)
                if (co < Cout) {
                    float o = acc[mt][nt][r];
                    if (scale)     o = o * scale[co] + (bias ? bias[co] : 0.0f);
                    else if (bias) o += bias[co];
                    if (relu && o < 0.0f) o = 0.0f;
                    if (out_pad) {
                        size_t oi = ((size_t)b * Cout + co) * PHW + (size_t)(h + 1) * PW + (n + 4);
                        outb[oi] = f2bf(o);
                    } else {
                        size_t oi = ((size_t)b * Cout + co) * HWC + (size_t)h * WW + n;
                        if (outf) outf[oi] = o;
                        else      outb[oi] = f2bf(o);
                    }
                }
            }
}

// -------------------- criss-cross attention: energies + softmax --------------------

__global__ __launch_bounds__(192)
void cca_att_k(const ushort_t* __restrict__ q, const ushort_t* __restrict__ k,
               float* __restrict__ att)
{
    __shared__ float qv[CQK];
    __shared__ float buf[192];
    __shared__ float stat[2];

    int s  = blockIdx.x;
    int b  = s / HWC;
    int hw = s % HWC;
    int h  = hw / WW, w = hw % WW;
    int t  = threadIdx.x;

    if (t < CQK) qv[t] = bf2f(q[((size_t)b * CQK + t) * HWC + hw]);
    __syncthreads();

    float e;
    if (t < HH) {
        int g = t;
        if (g == h) {
            e = -__builtin_inff();
        } else {
            float acc = 0.0f;
#pragma unroll 8
            for (int c = 0; c < CQK; ++c)
                acc += qv[c] * bf2f(k[((size_t)b * CQK + c) * HWC + g * WW + w]);
            e = acc;
        }
    } else {
        int vv = t - HH;
        float acc = 0.0f;
#pragma unroll 8
        for (int c = 0; c < CQK; ++c)
            acc += qv[c] * bf2f(k[((size_t)b * CQK + c) * HWC + h * WW + vv]);
        e = acc;
    }

    buf[t] = e;
    __syncthreads();
    if (t == 0) {
        float m = -__builtin_inff();
        for (int i = 0; i < 192; ++i) m = fmaxf(m, buf[i]);
        float den = 0.0f;
        for (int i = 0; i < 192; ++i) den += __expf(buf[i] - m);
        stat[0] = m; stat[1] = den;
    }
    __syncthreads();

    att[(size_t)s * 192 + t] = __expf(e - stat[0]) / stat[1];
}

// -------------------- criss-cross attention: aggregation + residual --------------------
// v unpadded; xin/out padded

__global__ __launch_bounds__(256)
void cca_agg_k(const ushort_t* __restrict__ v, const float* __restrict__ att,
               const ushort_t* __restrict__ xin, const float* __restrict__ gamma,
               ushort_t* __restrict__ out)
{
    int idx = blockIdx.x * blockDim.x + threadIdx.x;
    if (idx >= BB * CI * HWC) return;
    int w = idx % WW;
    int h = (idx / WW) % HH;
    int c = (idx / HWC) % CI;
    int b = idx / (CI * HWC);

    const float* a = &att[(((size_t)b * HWC) + h * WW + w) * 192];
    const ushort_t* vb = &v[((size_t)b * CI + c) * HWC];

    float sum = 0.0f;
#pragma unroll 8
    for (int g = 0; g < HH; ++g)
        sum += bf2f(vb[g * WW + w]) * a[g];
#pragma unroll 8
    for (int vv = 0; vv < WW; ++vv)
        sum += bf2f(vb[h * WW + vv]) * a[HH + vv];

    size_t pi = ((size_t)b * CI + c) * PHW + (size_t)(h + 1) * PW + (w + 4);
    float res = bf2f(xin[pi]);
    out[pi] = f2bf(gamma[0] * sum + res);
}

// -------------------- host-side launch --------------------

extern "C" void kernel_launch(void* const* d_in, const int* in_sizes, int n_in,
                              void* d_out, int out_size, void* d_ws, size_t ws_size,
                              hipStream_t stream) {
    (void)in_sizes; (void)n_in; (void)out_size; (void)ws_size;

    const float* x       = (const float*)d_in[0];
    const float* conva_w = (const float*)d_in[1];
    const float* bn1_s   = (const float*)d_in[2];
    const float* bn1_b   = (const float*)d_in[3];
    const float* q_w     = (const float*)d_in[4];
    const float* k_w     = (const float*)d_in[5];
    const float* v_w     = (const float*)d_in[6];
    const float* gamma   = (const float*)d_in[7];
    const float* convb_w = (const float*)d_in[8];
    const float* bn2_s   = (const float*)d_in[9];
    const float* bn2_b   = (const float*)d_in[10];
    const float* bott_w  = (const float*)d_in[11];
    const float* bn3_s   = (const float*)d_in[12];
    const float* bn3_b   = (const float*)d_in[13];
    const float* out_w   = (const float*)d_in[14];
    const float* out_b   = (const float*)d_in[15];

    char* ws = (char*)d_ws;
    size_t off = 0;
    auto alloc = [&](size_t bytes) -> void* {
        void* p = ws + off;
        off += (bytes + 255) & ~(size_t)255;
        return p;
    };

    ushort_t* xp     = (ushort_t*)alloc((size_t)BB * CIN * PHW * 2);   // padded input
    ushort_t* wa_t   = (ushort_t*)alloc((size_t)9 * CI * CIN * 2);
    ushort_t* wb_t   = (ushort_t*)alloc((size_t)9 * CI * CI * 2);
    ushort_t* wbt_t  = (ushort_t*)alloc((size_t)9 * CI * (CIN + CI) * 2);
    ushort_t* wq_t   = (ushort_t*)alloc((size_t)CQK * CI * 2);
    ushort_t* wk_t   = (ushort_t*)alloc((size_t)CQK * CI * 2);
    ushort_t* wv_t   = (ushort_t*)alloc((size_t)CI * CI * 2);
    ushort_t* wo_t   = (ushort_t*)alloc((size_t)NCLS * CI * 2);
    ushort_t* fa     = (ushort_t*)alloc((size_t)BB * CI * PHW * 2);    // padded feats
    ushort_t* fb     = (ushort_t*)alloc((size_t)BB * CI * PHW * 2);
    ushort_t* fc     = (ushort_t*)alloc((size_t)BB * CI * PHW * 2);
    ushort_t* qb     = (ushort_t*)alloc((size_t)BB * CQK * HWC * 2);   // unpadded
    ushort_t* kb     = (ushort_t*)alloc((size_t)BB * CQK * HWC * 2);
    ushort_t* vb     = (ushort_t*)alloc((size_t)BB * CI * HWC * 2);
    float*    att    = (float*)   alloc((size_t)BB * HWC * 192 * 4);

    // ---- padded cast of x, zero padded feature buffers ----
    {
        size_t n = (size_t)BB * CIN * PHW;
        cast_pad_k<<<(int)((n + 255) / 256), 256, 0, stream>>>(x, xp, CIN, n);
    }
    {
        size_t n = (size_t)BB * CI * PHW;
        zero_u16_k<<<2048, 256, 0, stream>>>(fa, n);
        zero_u16_k<<<2048, 256, 0, stream>>>(fb, n);
        zero_u16_k<<<2048, 256, 0, stream>>>(fc, n);
    }

    // ---- weight transposes: (Cout,Cin,Kpos) -> (Kpos,Cout,Cin) bf16 ----
    {
        int n = CI * CIN * 9;
        cast_transpose_w_k<<<(n + 255) / 256, 256, 0, stream>>>(conva_w, wa_t, CI, CIN, 9, n);
    }
    {
        int n = CI * CI * 9;
        cast_transpose_w_k<<<(n + 255) / 256, 256, 0, stream>>>(convb_w, wb_t, CI, CI, 9, n);
    }
    {
        int n = CI * (CIN + CI) * 9;
        cast_transpose_w_k<<<(n + 255) / 256, 256, 0, stream>>>(bott_w, wbt_t, CI, CIN + CI, 9, n);
    }
    {
        int n = CQK * CI;
        cast_transpose_w_k<<<(n + 255) / 256, 256, 0, stream>>>(q_w, wq_t, CQK, CI, 1, n);
        cast_transpose_w_k<<<(n + 255) / 256, 256, 0, stream>>>(k_w, wk_t, CQK, CI, 1, n);
    }
    {
        int n = CI * CI;
        cast_transpose_w_k<<<(n + 255) / 256, 256, 0, stream>>>(v_w, wv_t, CI, CI, 1, n);
    }
    {
        int n = NCLS * CI;
        cast_transpose_w_k<<<(n + 255) / 256, 256, 0, stream>>>(out_w, wo_t, NCLS, CI, 1, n);
    }

    const dim3 blk(256);
    const dim3 gRows(BB * HH, CI / BM);   // (192, 4)
    const dim3 gRows1(BB * HH, 1);        // Cout <= 128

    // ---- conva + bn1 + relu  -> fa (padded) ----
    conv_gemm_wmma_k<9><<<gRows, blk, 0, stream>>>(xp, CIN, nullptr, 0, wa_t, CI,
                                                   bn1_s, bn1_b, 1, 1, fa, nullptr);

    // ---- criss-cross attention x2 (fa -> fb -> fa) ----
    ushort_t* cur = fa;
    ushort_t* nxt = fb;
    for (int rec = 0; rec < 2; ++rec) {
        conv_gemm_wmma_k<1><<<gRows1, blk, 0, stream>>>(cur, CI, nullptr, 0, wq_t, CQK,
                                                        nullptr, nullptr, 0, 0, qb, nullptr);
        conv_gemm_wmma_k<1><<<gRows1, blk, 0, stream>>>(cur, CI, nullptr, 0, wk_t, CQK,
                                                        nullptr, nullptr, 0, 0, kb, nullptr);
        conv_gemm_wmma_k<1><<<gRows, blk, 0, stream>>>(cur, CI, nullptr, 0, wv_t, CI,
                                                       nullptr, nullptr, 0, 0, vb, nullptr);
        cca_att_k<<<BHW, 192, 0, stream>>>(qb, kb, att);
        {
            int n = BB * CI * HWC;
            cca_agg_k<<<(n + 255) / 256, 256, 0, stream>>>(vb, att, cur, gamma, nxt);
        }
        ushort_t* tmp = cur; cur = nxt; nxt = tmp;
    }
    // cur == fa

    // ---- convb + bn2 + relu (fa -> fb, padded) ----
    conv_gemm_wmma_k<9><<<gRows, blk, 0, stream>>>(cur, CI, nullptr, 0, wb_t, CI,
                                                   bn2_s, bn2_b, 1, 1, fb, nullptr);

    // ---- bottleneck conv on virtual concat [x(2048) | fb(512)] + bn3 -> fc (padded) ----
    conv_gemm_wmma_k<9><<<gRows, blk, 0, stream>>>(xp, CIN, fb, CI, wbt_t, CI,
                                                   bn3_s, bn3_b, 0, 1, fc, nullptr);

    // ---- classifier 1x1: 512 -> 19, + bias, f32 unpadded -> d_out ----
    conv_gemm_wmma_k<1><<<gRows1, blk, 0, stream>>>(fc, CI, nullptr, 0, wo_t, NCLS,
                                                    nullptr, out_b, 0, 0, nullptr, (float*)d_out);
}
